// SimpleEuclideanCodebook_35467839930394
// MI455X (gfx1250) — matrix operand order repaired and checked by reading, not compiled, not run on hardware
//
#include <hip/hip_runtime.h>
#include <hip/hip_bf16.h>

typedef __attribute__((ext_vector_type(16))) __bf16 v16bf;
typedef __attribute__((ext_vector_type(8)))  float  v8f;
typedef __attribute__((__vector_size__(16))) int    v4i_t;   // async builtin elem type

#define DIMS 256
#define ROWS_PER_BLOCK 128
#define NWAVES 8
#define CHUNK 64                 // codewords staged in LDS per iteration
#define NCHUNKS 16               // 1024 / CHUNK
#define LDS_STRIDE 272           // 256 + 16 halves pad -> 8-dword bank rotation per row
#define GROUPS_PER_CHUNK ((CHUNK * DIMS) / (256 * 8))   // b128 groups per thread = 8

#define AS1 __attribute__((address_space(1)))
#define AS3 __attribute__((address_space(3)))

#if defined(__has_builtin)
#  if __has_builtin(__builtin_amdgcn_global_load_async_to_lds_b128)
#    define HAVE_ASYNC_LDS 1
#  endif
#  if __has_builtin(__builtin_amdgcn_s_wait_asynccnt)
#    define WAIT_ASYNC(n) __builtin_amdgcn_s_wait_asynccnt(n)
#  endif
#endif
#ifndef HAVE_ASYNC_LDS
#  define HAVE_ASYNC_LDS 0
#endif
#ifndef WAIT_ASYNC
#  define WAIT_ASYNC(n) asm volatile("s_wait_asynccnt %0" ::"i"(n) : "memory")
#endif

static __device__ __forceinline__ unsigned short f2bf(float f) {
    unsigned u = __float_as_uint(f);
    u += 0x7fffu + ((u >> 16) & 1u);     // round-to-nearest-even
    return (unsigned short)(u >> 16);
}
static __device__ __forceinline__ unsigned pk2(float lo, float hi) {
    return (unsigned)f2bf(lo) | ((unsigned)f2bf(hi) << 16);
}

union Frag { v16bf v; unsigned u[8]; };

// Stage one CHUNK of bf16 codewords into a (padded-stride) LDS buffer.
// Async-to-LDS on gfx1250 (ASYNCcnt-tracked), synchronous copy as fallback.
static __device__ __forceinline__
void stage_chunk(unsigned short* dstbuf, const unsigned short* __restrict__ embed_bf,
                 int chunkBase, int tid) {
    #pragma unroll
    for (int it = 0; it < GROUPS_PER_CHUNK; ++it) {
        int g = it * 256 + tid;
        int r = g >> 5;                   // 32 groups of 8 halves per row
        int c = (g & 31) << 3;
        const unsigned short* src = &embed_bf[(size_t)(chunkBase + r) * DIMS + c];
        unsigned short* dst = &dstbuf[r * LDS_STRIDE + c];
#if HAVE_ASYNC_LDS
        __builtin_amdgcn_global_load_async_to_lds_b128(
            (AS1 v4i_t*)src, (AS3 v4i_t*)dst, 0, 0);
#else
        *(uint4*)dst = *(const uint4*)src;
#endif
    }
}

// ---- prep: embed fp32 -> bf16 table + (-0.5 * ||e||^2) per codeword ----
__global__ void __launch_bounds__(DIMS)
vq_prep_kernel(const float* __restrict__ embed,
               unsigned short* __restrict__ embed_bf,
               float* __restrict__ negHalf) {
    int b = blockIdx.x, t = threadIdx.x;
    float v = embed[(size_t)b * DIMS + t];
    embed_bf[(size_t)b * DIMS + t] = f2bf(v);
    float s = v * v;
    #pragma unroll
    for (int m = 16; m >= 1; m >>= 1) s += __shfl_xor(s, m, 32);
    __shared__ float p[NWAVES];
    if ((t & 31) == 0) p[t >> 5] = s;
    __syncthreads();
    if (t == 0) {
        float tot = 0.f;
        #pragma unroll
        for (int i = 0; i < NWAVES; ++i) tot += p[i];
        negHalf[b] = -0.5f * tot;
    }
}

// ---- main: fused score GEMM (bf16 WMMA) + row argmax + fp32 gather ----
__global__ void __launch_bounds__(256)
vq_main_kernel(const float* __restrict__ x,
               const float* __restrict__ embed,
               const unsigned short* __restrict__ embed_bf,
               const float* __restrict__ negHalf,
               float* __restrict__ q_out,
               float* __restrict__ idx_out) {
    __shared__ unsigned short lds_b[2][CHUNK * LDS_STRIDE];   // double buffer
    __shared__ int idx_lds[ROWS_PER_BLOCK];

    const int tid    = threadIdx.x;
    const int lane   = tid & 31;
    const int wave   = tid >> 5;
    const int l15    = lane & 15;
    const int hiHalf = (lane >> 4) & 1;      // 0: lanes 0-15, 1: lanes 16-31
    const int akoff  = hiHalf * 8;           // A: interleaved K groups

    const int row0 = blockIdx.x * ROWS_PER_BLOCK + wave * 16;
    const int arow = row0 + l15;

    // Load & convert the wave's A tile (16 rows x 256 K) into registers.
    // A layout (16-bit): lanes 0-15 hold K {0..7,16..23}, lanes 16-31 hold {8..15,24..31}.
    Frag a[8];
    #pragma unroll
    for (int ks = 0; ks < 8; ++ks) {
        const float* xp = x + (size_t)arow * DIMS + ks * 32 + akoff;
        float4 f0 = *(const float4*)(xp + 0);
        float4 f1 = *(const float4*)(xp + 4);
        float4 f2 = *(const float4*)(xp + 16);
        float4 f3 = *(const float4*)(xp + 20);
        a[ks].u[0] = pk2(f0.x, f0.y);  a[ks].u[1] = pk2(f0.z, f0.w);
        a[ks].u[2] = pk2(f1.x, f1.y);  a[ks].u[3] = pk2(f1.z, f1.w);
        a[ks].u[4] = pk2(f2.x, f2.y);  a[ks].u[5] = pk2(f2.z, f2.w);
        a[ks].u[6] = pk2(f3.x, f3.y);  a[ks].u[7] = pk2(f3.z, f3.w);
    }

    float best[8]; int bidx[8];
    #pragma unroll
    for (int i = 0; i < 8; ++i) { best[i] = -3.4e38f; bidx[i] = 0; }

    // Prologue: kick off chunk 0 into buffer 0.
    stage_chunk(&lds_b[0][0], embed_bf, 0, tid);

    for (int ch = 0; ch < NCHUNKS; ++ch) {
        const int cur = ch & 1;
        __syncthreads();                 // readers of buf[cur^1] are done with it
        if (ch + 1 < NCHUNKS)            // pipeline: fetch next chunk under compute
            stage_chunk(&lds_b[cur ^ 1][0], embed_bf, (ch + 1) * CHUNK, tid);
#if HAVE_ASYNC_LDS
        // Async loads complete in order: waiting for <= GROUPS_PER_CHUNK leaves
        // only the just-issued prefetch outstanding, so buf[cur] is complete.
        if (ch + 1 < NCHUNKS) WAIT_ASYNC(GROUPS_PER_CHUNK);
        else                  WAIT_ASYNC(0);
#endif
        __syncthreads();                 // buf[cur] visible to all waves

        const unsigned short* bbuf = &lds_b[cur][0];
        #pragma unroll
        for (int ct = 0; ct < CHUNK / 16; ++ct) {
            const int cw = ch * CHUNK + ct * 16 + l15;       // this lane's column
            const float cinit = negHalf[cw];                 // -0.5*||e||^2 (L2 hit)
            v8f acc = {0.f, 0.f, 0.f, 0.f, 0.f, 0.f, 0.f, 0.f};  // SRC2 inline 0
            #pragma unroll
            for (int ks = 0; ks < 8; ++ks) {
                // B layout (16-bit, 32x16): lanes 0-15 K={0..7,16..23}, lanes 16-31 K={8..15,24..31}
                const unsigned short* bp =
                    &bbuf[(ct * 16 + l15) * LDS_STRIDE + ks * 32 + hiHalf * 8];
                Frag b;
                *(uint4*)&b.u[0] = *(const uint4*)(bp);
                *(uint4*)&b.u[4] = *(const uint4*)(bp + 16);
                acc = __builtin_amdgcn_wmma_f32_16x16x32_bf16(
                        false, a[ks].v, false, b.v, (short)0, acc, false, false);
            }
            #pragma unroll
            for (int i = 0; i < 8; ++i) {
                float s = acc[i] + cinit;                    // apply bias off-chain
                if (s > best[i]) { best[i] = s; bidx[i] = cw; }
            }
        }
    }

    // Cross-lane argmax within each 16-lane half (xor masks 1..8 stay in-half).
    // Tie-break: lower index (matches jnp.argmax first-occurrence).
    #pragma unroll
    for (int m = 1; m <= 8; m <<= 1) {
        #pragma unroll
        for (int i = 0; i < 8; ++i) {
            float os = __shfl_xor(best[i], m, 32);
            int   oi = __shfl_xor(bidx[i], m, 32);
            if (os > best[i] || (os == best[i] && oi < bidx[i])) {
                best[i] = os; bidx[i] = oi;
            }
        }
    }

    // C/D layout: lanes 0-15 rows row0+0..7 (reg i -> M=i), lanes 16-31 rows row0+8..15.
    if (l15 == 0) {
        int lbase = wave * 16 + hiHalf * 8;
        #pragma unroll
        for (int i = 0; i < 8; ++i) {
            idx_lds[lbase + i] = bidx[i];
            idx_out[row0 + hiHalf * 8 + i] = (float)bidx[i];
        }
    }
    __syncthreads();

    // Cooperative fp32 gather: 128 rows x 256 floats from the exact codebook.
    const size_t out_base = (size_t)blockIdx.x * ROWS_PER_BLOCK * DIMS;
    #pragma unroll
    for (int it = 0; it < (ROWS_PER_BLOCK * DIMS / 4) / 256; ++it) {
        int g = it * 256 + tid;
        int r = g >> 6;                  // 64 float4 per row
        int c = (g & 63) << 2;
        int cw = idx_lds[r];
        *(float4*)&q_out[out_base + (size_t)r * DIMS + c] =
            *(const float4*)&embed[(size_t)cw * DIMS + c];
    }
}

extern "C" void kernel_launch(void* const* d_in, const int* in_sizes, int n_in,
                              void* d_out, int out_size, void* d_ws, size_t ws_size,
                              hipStream_t stream) {
    const float* x     = (const float*)d_in[0];
    const float* embed = (const float*)d_in[1];
    const int N = in_sizes[0] / DIMS;      // 32768 rows
    const int K = in_sizes[1] / DIMS;      // 1024 codewords

    unsigned short* embed_bf = (unsigned short*)d_ws;
    float* negHalf = (float*)((char*)d_ws + (size_t)K * DIMS * sizeof(unsigned short));

    float* q_out   = (float*)d_out;
    float* idx_out = q_out + (size_t)N * DIMS;

    vq_prep_kernel<<<K, DIMS, 0, stream>>>(embed, embed_bf, negHalf);
    vq_main_kernel<<<N / ROWS_PER_BLOCK, 256, 0, stream>>>(
        x, embed, embed_bf, negHalf, q_out, idx_out);
}
